// EPTLayer_79963701117013
// MI455X (gfx1250) — matrix operand order repaired and checked by reading, not compiled, not run on hardware
//
#include <hip/hip_runtime.h>

// ---------------------------------------------------------------------------
// Problem constants (from reference)
// ---------------------------------------------------------------------------
constexpr int cB     = 2;
constexpr int cN     = 1024;
constexpr int cDHID  = 512;
constexpr int cNH    = 16;
constexpr int cBN    = cB * cN;        // 2048
constexpr int cBN3   = cBN * 3;        // 6144
#define FACTOR 0.08838834764831845f    // 0.5 / sqrt(32)

typedef unsigned short u16;
typedef __attribute__((ext_vector_type(16))) __bf16 v16bf;
typedef __attribute__((ext_vector_type(8)))  __bf16 v8bf;
typedef __attribute__((ext_vector_type(8)))  float  v8f;

__device__ __forceinline__ u16 f2b(float f) {
  union { float f; unsigned u; } x; x.f = f;
  unsigned u = x.u;
  return (u16)((u + 0x7FFFu + ((u >> 16) & 1u)) >> 16);   // RNE bf16
}

// Fragment load: two aligned 16B LDS chunks -> one 16-elem bf16 fragment.
__device__ __forceinline__ v16bf ld_frag(const u16* p0, const u16* p1) {
  v8bf a = *(const v8bf*)p0;
  v8bf b = *(const v8bf*)p1;
  return __builtin_shufflevector(a, b, 0,1,2,3,4,5,6,7,8,9,10,11,12,13,14,15);
}

// Async global->LDS 16B copy (GLOBAL_LOAD_ASYNC_TO_LDS_B128, ASYNCcnt).
__device__ __forceinline__ void async_b128(void* lds, const void* gptr) {
  unsigned lo = (unsigned)(size_t)lds;       // LDS offset = low 32 bits
  asm volatile("global_load_async_to_lds_b128 %0, %1, off"
               :: "v"(lo), "v"(gptr) : "memory");
}
__device__ __forceinline__ void wait_async0() {
  asm volatile("s_wait_asynccnt 0x0" ::: "memory");
}

// ---------------------------------------------------------------------------
// f32 -> bf16 (plain) and f32 -> bf16 transposed (weights: Wt[n][k]=W[k][n])
// ---------------------------------------------------------------------------
__global__ void cvt_kernel(const float* __restrict__ in, u16* __restrict__ out, int n) {
  for (int i = blockIdx.x * blockDim.x + threadIdx.x; i < n; i += gridDim.x * blockDim.x)
    out[i] = f2b(in[i]);
}
__global__ void cvt_t_kernel(const float* __restrict__ in, u16* __restrict__ out,
                             int K, int Nc) {
  int total = K * Nc;
  for (int i = blockIdx.x * blockDim.x + threadIdx.x; i < total; i += gridDim.x * blockDim.x) {
    int k = i / Nc, n = i % Nc;
    out[(size_t)n * K + k] = f2b(in[i]);
  }
}

// ---------------------------------------------------------------------------
// LayerNorm over 512 cols, one wave32 per row, bf16 output
// ---------------------------------------------------------------------------
__global__ __launch_bounds__(128) void ln_kernel(const float* __restrict__ X,
                                                 const float* __restrict__ g,
                                                 const float* __restrict__ bta,
                                                 u16* __restrict__ out, int rows) {
  int w = threadIdx.x >> 5, lane = threadIdx.x & 31;
  int row = blockIdx.x * 4 + w;
  if (row >= rows) return;
  const float* x = X + (size_t)row * cDHID;
  float v[16], s = 0.f, s2 = 0.f;
#pragma unroll
  for (int i = 0; i < 16; ++i) { float t = x[lane + i * 32]; v[i] = t; s += t; s2 += t * t; }
#pragma unroll
  for (int m = 1; m < 32; m <<= 1) { s += __shfl_xor(s, m, 32); s2 += __shfl_xor(s2, m, 32); }
  float mean = s * (1.f / cDHID);
  float var  = s2 * (1.f / cDHID) - mean * mean;
  float rstd = rsqrtf(var + 1e-5f);
#pragma unroll
  for (int i = 0; i < 16; ++i) {
    int c = lane + i * 32;
    out[(size_t)row * cDHID + c] = f2b((v[i] - mean) * rstd * g[c] + bta[c]);
  }
}

// ---------------------------------------------------------------------------
// bf16 WMMA GEMM, double-buffered async staging.
// C[M,Nc] = A[M,K] * Wt[Nc,K]^T (+bias)(+resid)(+silu)
// Tile 128x64, BK=32; 8 waves, each 32x32 (2x2 16x16x32 bf16 frags).
// ---------------------------------------------------------------------------
#define GTM 128
#define GTN 64
#define GTK 32
#define ASTR 40   // LDS row stride (elems); 80B, 16B-aligned
#define BSTR 40

__global__ __launch_bounds__(256) void gemm_bf16_kernel(
    const u16* __restrict__ A, const u16* __restrict__ Wt,
    const float* __restrict__ bias, const float* __restrict__ resid,
    float* __restrict__ outF, u16* __restrict__ outB,
    int M, int K, int Nc, int act) {
  __shared__ u16 As[2][GTM][ASTR];   // 20480 B
  __shared__ u16 Bts[2][GTN][BSTR];  // 10240 B (B tile stored K-contiguous)

  const int tid  = threadIdx.x;
  const int lane = tid & 31;
  const int w    = tid >> 5;
  const int wm   = w & 3;
  const int wn   = w >> 2;
  const int r    = lane & 15;
  const int hi   = lane >> 4;
  const int m0   = blockIdx.y * GTM;
  const int n0   = blockIdx.x * GTN;

  const int a_row = tid >> 2, a_c = tid & 3;   // A: 512 chunks (2/thread)
  const int b_row = tid >> 2, b_c = tid & 3;   // B: 256 chunks (1/thread)

  auto stage = [&](int k0, int buf) {
#pragma unroll
    for (int it = 0; it < 2; ++it) {
      int row = a_row + it * 64;
      async_b128(&As[buf][row][a_c * 8],
                 &A[(size_t)(m0 + row) * K + k0 + a_c * 8]);
    }
    async_b128(&Bts[buf][b_row][b_c * 8],
               &Wt[(size_t)(n0 + b_row) * K + k0 + b_c * 8]);
  };

  v8f acc[2][2];
#pragma unroll
  for (int mi = 0; mi < 2; ++mi)
#pragma unroll
    for (int ni = 0; ni < 2; ++ni)
      acc[mi][ni] = (v8f){0.f,0.f,0.f,0.f,0.f,0.f,0.f,0.f};

  stage(0, 0);
  wait_async0();
  __syncthreads();

  int cur = 0;
  for (int k0 = 0; k0 < K; k0 += GTK) {
    if (k0 + GTK < K) stage(k0 + GTK, cur ^ 1);

    v16bf afr[2], bfr[2];
#pragma unroll
    for (int mi = 0; mi < 2; ++mi) {
      const u16* p = &As[cur][wm * 32 + mi * 16 + r][0];
      afr[mi] = ld_frag(p + hi * 8, p + 16 + hi * 8);
    }
#pragma unroll
    for (int ni = 0; ni < 2; ++ni) {
      const u16* p = &Bts[cur][wn * 32 + ni * 16 + r][0];
      bfr[ni] = ld_frag(p + hi * 8, p + 16 + hi * 8);
    }
#pragma unroll
    for (int mi = 0; mi < 2; ++mi)
#pragma unroll
      for (int ni = 0; ni < 2; ++ni)
        acc[mi][ni] = __builtin_amdgcn_wmma_f32_16x16x32_bf16(
            false, afr[mi], false, bfr[ni], (short)0, acc[mi][ni], false, false);

    wait_async0();
    __syncthreads();
    cur ^= 1;
  }

  // epilogue
#pragma unroll
  for (int mi = 0; mi < 2; ++mi) {
#pragma unroll
    for (int ni = 0; ni < 2; ++ni) {
      int rowb = m0 + wm * 32 + mi * 16 + hi * 8;
      int col  = n0 + wn * 32 + ni * 16 + r;
      float bv = bias ? bias[col] : 0.f;
#pragma unroll
      for (int i = 0; i < 8; ++i) {
        size_t off = (size_t)(rowb + i) * Nc + col;
        float v = acc[mi][ni][i] + bv;
        if (resid) v += resid[off];
        if (act == 1) v = v / (1.f + __expf(-v));   // silu
        if (outF) outF[off] = v;
        if (outB) outB[off] = f2b(v);
      }
    }
  }
}

// ---------------------------------------------------------------------------
// Pack V_attn[bn, h*128 + d]: d<32 from Hv; else from Vv (head-interleave)
// ---------------------------------------------------------------------------
__global__ void pack_vattn_kernel(const u16* __restrict__ hv, const u16* __restrict__ vv,
                                  u16* __restrict__ vat) {
  int total = cBN * 2048;
  for (int i = blockIdx.x * blockDim.x + threadIdx.x; i < total; i += gridDim.x * blockDim.x) {
    int bn = i >> 11, d2 = i & 2047;
    int h = d2 >> 7, d = d2 & 127;
    u16 val;
    if (d < 32) val = hv[(size_t)bn * cDHID + h * 32 + d];
    else { int c = (d - 32) >> 5, dd = (d - 32) & 31;
           val = vv[((size_t)bn * 3 + c) * cDHID + h * 32 + dd]; }
    vat[i] = val;
  }
}

// ---------------------------------------------------------------------------
// Flash attention. Block = 32 queries x one (b,h); 64-key tiles.
// Q/K staged via async b128 (row-major, K-contiguous => direct B frags for
// S = Q K^T). V staged transposed in LDS (d x key) => direct B frags for
// O += P V. 128 threads = 4 waves.
// ---------------------------------------------------------------------------
#define QSTR 136   // 128+8 elems, 272B (16B mult)
#define VSTR 72    // 64+8 elems, 144B (16B mult)

__global__ __launch_bounds__(128) void attn_kernel(
    const u16* __restrict__ Q, const u16* __restrict__ Kb,
    const u16* __restrict__ Vat, const float* __restrict__ rbf,
    const float* __restrict__ Dm, const int* __restrict__ mask,
    u16* __restrict__ resH, u16* __restrict__ resV) {
  __shared__ u16 Qs[32][QSTR];       //  8704 B
  __shared__ u16 Ks[64][QSTR];       // 17408 B
  __shared__ u16 Vt[128][VSTR];      // 18432 B (transposed: [d][key])
  __shared__ u16 Ps[32][VSTR];       //  4608 B
  __shared__ float m_run[32], l_run[32], alpha_s[32];
  __shared__ float pmax[32][2], psum[32][2];

  const int tid  = threadIdx.x;
  const int lane = tid & 31, w = tid >> 5;
  const int r = lane & 15, hi = lane >> 4;
  const int wq = w & 1, wk = w >> 1;
  const int wn = w >> 1;
  const int q0 = blockIdx.x * 32;
  const int h  = blockIdx.y;
  const int b  = blockIdx.z;

  // Q tile: 32x128 bf16 = 512 x 16B chunks, async
#pragma unroll
  for (int it = 0; it < 4; ++it) {
    int chunk = tid + it * 128;
    int row = chunk >> 4, c = chunk & 15;
    async_b128(&Qs[row][c * 8],
               &Q[(size_t)(b * cN + q0 + row) * 2048 + h * 128 + c * 8]);
  }
  if (tid < 32) { m_run[tid] = -1e30f; l_run[tid] = 0.f; }

  v8f o[4];
#pragma unroll
  for (int ci = 0; ci < 4; ++ci) o[ci] = (v8f){0.f,0.f,0.f,0.f,0.f,0.f,0.f,0.f};

  for (int kt = 0; kt < cN / 64; ++kt) {
    const int k0 = kt * 64;
    __syncthreads();   // prior consumers of Ks/Vt/Ps done
    // K tile: 64x128 = 1024 chunks, async
#pragma unroll
    for (int it = 0; it < 8; ++it) {
      int chunk = tid + it * 128;
      int row = chunk >> 4, c = chunk & 15;
      async_b128(&Ks[row][c * 8],
                 &Kb[(size_t)(b * cN + k0 + row) * 2048 + h * 128 + c * 8]);
    }
    // V tile transposed: read b128, scatter 8 ushorts into Vt[d][key]
#pragma unroll
    for (int it = 0; it < 8; ++it) {
      int chunk = tid + it * 128;
      int key = chunk >> 4, c = chunk & 15;
      union { uint4 q; u16 s[8]; } u;
      u.q = *(const uint4*)&Vat[(size_t)(b * cN + k0 + key) * 2048 + h * 128 + c * 8];
#pragma unroll
      for (int j = 0; j < 8; ++j) Vt[c * 8 + j][key] = u.s[j];
    }
    wait_async0();
    __syncthreads();

    // ---- S = Q K^T ----
    v8f s[2];
    s[0] = (v8f){0.f,0.f,0.f,0.f,0.f,0.f,0.f,0.f};
    s[1] = (v8f){0.f,0.f,0.f,0.f,0.f,0.f,0.f,0.f};
#pragma unroll
    for (int d0 = 0; d0 < 128; d0 += 32) {
      const u16* qp = &Qs[wq * 16 + r][d0];
      v16bf a = ld_frag(qp + hi * 8, qp + 16 + hi * 8);
#pragma unroll
      for (int ni = 0; ni < 2; ++ni) {
        const u16* kp = &Ks[wk * 32 + ni * 16 + r][d0];
        v16bf bb = ld_frag(kp + hi * 8, kp + 16 + hi * 8);
        s[ni] = __builtin_amdgcn_wmma_f32_16x16x32_bf16(
            false, a, false, bb, (short)0, s[ni], false, false);
      }
    }

    // ---- bias + mask, row-max ----
    float sv[2][8];
#pragma unroll
    for (int ni = 0; ni < 2; ++ni) {
#pragma unroll
      for (int i = 0; i < 8; ++i) {
        int qg = q0 + wq * 16 + hi * 8 + i;
        int kg = k0 + wk * 32 + ni * 16 + r;
        float val = s[ni][i] * FACTOR
                  + rbf[(((size_t)b * cNH + h) * cN + qg) * cN + kg]
                  + Dm[((size_t)b * cN + qg) * cN + kg];
        if (mask[b * cN + kg] == 0) val = -1e30f;
        sv[ni][i] = val;
      }
    }
#pragma unroll
    for (int i = 0; i < 8; ++i) {
      float vm = fmaxf(sv[0][i], sv[1][i]);
#pragma unroll
      for (int m = 1; m < 16; m <<= 1) vm = fmaxf(vm, __shfl_xor(vm, m, 32));
      if (r == 0) pmax[wq * 16 + hi * 8 + i][wk] = vm;
    }
    __syncthreads();
    if (tid < 32) {
      float mt = fmaxf(pmax[tid][0], pmax[tid][1]);
      float mn = fmaxf(m_run[tid], mt);
      alpha_s[tid] = __expf(m_run[tid] - mn);
      m_run[tid]   = mn;
    }
    __syncthreads();

    // ---- P = exp(sv - m), bf16 P tile, partial row sums ----
#pragma unroll
    for (int i = 0; i < 8; ++i) {
      int prow = wq * 16 + hi * 8 + i;
      float mrow = m_run[prow];
      float p0 = __expf(sv[0][i] - mrow);
      float p1 = __expf(sv[1][i] - mrow);
      Ps[prow][wk * 32 + r]      = f2b(p0);
      Ps[prow][wk * 32 + 16 + r] = f2b(p1);
      float ps = p0 + p1;
#pragma unroll
      for (int m = 1; m < 16; m <<= 1) ps += __shfl_xor(ps, m, 32);
      if (r == 0) psum[prow][wk] = ps;
    }
    __syncthreads();
    if (tid < 32) l_run[tid] = l_run[tid] * alpha_s[tid] + psum[tid][0] + psum[tid][1];

    // ---- O = O*alpha + P @ V ----
#pragma unroll
    for (int i = 0; i < 8; ++i) {
      float al = alpha_s[wq * 16 + hi * 8 + i];
#pragma unroll
      for (int ci = 0; ci < 4; ++ci) o[ci][i] *= al;
    }
#pragma unroll
    for (int ks = 0; ks < 64; ks += 32) {
      const u16* pp = &Ps[wq * 16 + r][ks];
      v16bf a = ld_frag(pp + hi * 8, pp + 16 + hi * 8);
#pragma unroll
      for (int ci = 0; ci < 4; ++ci) {
        const u16* vp = &Vt[wn * 64 + ci * 16 + r][ks];
        v16bf bb = ld_frag(vp + hi * 8, vp + 16 + hi * 8);
        o[ci] = __builtin_amdgcn_wmma_f32_16x16x32_bf16(
            false, a, false, bb, (short)0, o[ci], false, false);
      }
    }
  }
  __syncthreads();

  // ---- normalize + scatter into resH / resV ----
#pragma unroll
  for (int i = 0; i < 8; ++i) {
    float l = l_run[wq * 16 + hi * 8 + i];
    float linv = (l > 0.f) ? 1.f / l : 0.f;
    int qg = q0 + wq * 16 + hi * 8 + i;
#pragma unroll
    for (int ci = 0; ci < 4; ++ci) {
      int dfull = wn * 64 + ci * 16 + r;
      u16 hv = f2b(o[ci][i] * linv);
      if (dfull < 32) {
        resH[((size_t)(b * cN + qg)) * cDHID + h * 32 + dfull] = hv;
      } else {
        int c = (dfull - 32) >> 5, dd = (dfull - 32) & 31;
        resV[(((size_t)(b * cN + qg)) * 3 + c) * cDHID + h * 32 + dd] = hv;
      }
    }
  }
}

// ---------------------------------------------------------------------------
// scaler = concat(Hn2, ||V1||_{axis=c}) bf16; V1 = Vp[..., :512]
// ---------------------------------------------------------------------------
__global__ void scaler_kernel(const u16* __restrict__ hn2, const float* __restrict__ vp,
                              u16* __restrict__ scl) {
  int total = cBN * 1024;
  for (int i = blockIdx.x * blockDim.x + threadIdx.x; i < total; i += gridDim.x * blockDim.x) {
    int bn = i >> 10, j = i & 1023;
    if (j < cDHID) {
      scl[i] = hn2[(size_t)bn * cDHID + j];
    } else {
      int jj = j - cDHID;
      float a = vp[((size_t)bn * 3 + 0) * 1024 + jj];
      float c = vp[((size_t)bn * 3 + 1) * 1024 + jj];
      float d = vp[((size_t)bn * 3 + 2) * 1024 + jj];
      scl[i] = f2b(sqrtf(a * a + c * c + d * d));
    }
  }
}

// ---------------------------------------------------------------------------
// final: H_out = H_mid + s[:,:512];  V_out = V_mid + s[:,512:,None]*V2
// ---------------------------------------------------------------------------
__global__ void final_kernel(const float* __restrict__ hmid, const float* __restrict__ vmid,
                             const float* __restrict__ s, const float* __restrict__ vp,
                             float* __restrict__ out) {
  const int HN = cBN * cDHID;
  const int VN = cBN3 * cDHID;
  for (int i = blockIdx.x * blockDim.x + threadIdx.x; i < HN + VN;
       i += gridDim.x * blockDim.x) {
    if (i < HN) {
      int bn = i >> 9, j = i & 511;
      out[i] = hmid[i] + s[(size_t)bn * 1024 + j];
    } else {
      int t = i - HN;
      int row = t >> 9, j = t & 511;     // row = bn*3 + c
      int bn = row / 3;
      out[HN + t] = vmid[t] + s[(size_t)bn * 1024 + 512 + j]
                            * vp[(size_t)row * 1024 + 512 + j];
    }
  }
}

// ---------------------------------------------------------------------------
// host
// ---------------------------------------------------------------------------
extern "C" void kernel_launch(void* const* d_in, const int* in_sizes, int n_in,
                              void* d_out, int out_size, void* d_ws, size_t ws_size,
                              hipStream_t stream) {
  (void)in_sizes; (void)n_in; (void)out_size; (void)ws_size;
  const float* H    = (const float*)d_in[0];
  const float* Vin  = (const float*)d_in[1];
  const float* Dm   = (const float*)d_in[2];
  const float* rbf  = (const float*)d_in[3];
  const int*   mask = (const int*)d_in[4];
  const float* Wq   = (const float*)d_in[5];
  const float* bq   = (const float*)d_in[6];
  const float* Wk   = (const float*)d_in[7];
  const float* bk   = (const float*)d_in[8];
  const float* Wvs  = (const float*)d_in[9];
  const float* bvs  = (const float*)d_in[10];
  const float* Wvv  = (const float*)d_in[11];
  const float* Wo   = (const float*)d_in[12];
  const float* bo   = (const float*)d_in[13];
  const float* Wvo  = (const float*)d_in[14];
  const float* ln1g = (const float*)d_in[15];
  const float* ln1b = (const float*)d_in[16];
  const float* Wlv  = (const float*)d_in[17];
  const float* W1   = (const float*)d_in[18];
  const float* b1   = (const float*)d_in[19];
  const float* W2   = (const float*)d_in[20];
  const float* b2   = (const float*)d_in[21];
  const float* ln2g = (const float*)d_in[22];
  const float* ln2b = (const float*)d_in[23];

  char* ws = (char*)d_ws;
  auto alloc = [&](size_t bytes) -> char* {
    char* p = ws; ws += (bytes + 255) & ~(size_t)255; return p;
  };
  u16* wq_b   = (u16*)alloc((size_t)512 * 2048 * 2);   // transposed bf16 weights
  u16* wk_b   = (u16*)alloc((size_t)512 * 2048 * 2);
  u16* wvs_b  = (u16*)alloc((size_t)512 * 512 * 2);
  u16* wvv_b  = (u16*)alloc((size_t)512 * 512 * 2);
  u16* wo_b   = (u16*)alloc((size_t)512 * 512 * 2);
  u16* wvo_b  = (u16*)alloc((size_t)512 * 512 * 2);
  u16* wlv_b  = (u16*)alloc((size_t)512 * 1024 * 2);
  u16* w1_b   = (u16*)alloc((size_t)1024 * 2048 * 2);
  u16* w2_b   = (u16*)alloc((size_t)2048 * 1024 * 2);
  u16* vin_b  = (u16*)alloc((size_t)cBN3 * 512 * 2);
  u16* hn_b   = (u16*)alloc((size_t)cBN * 512 * 2);
  u16* q_b    = (u16*)alloc((size_t)cBN * 2048 * 2);   // }
  u16* k_b    = (u16*)alloc((size_t)cBN * 2048 * 2);   // } aliased by vp_f
  u16* vatt_b = (u16*)alloc((size_t)cBN * 2048 * 2);   // }
  u16* hv_b   = (u16*)alloc((size_t)cBN * 512 * 2);    // } aliased by hid_b
  u16* vv_b   = (u16*)alloc((size_t)cBN3 * 512 * 2);   // }
  u16* resH_b = (u16*)alloc((size_t)cBN * 512 * 2);    // } aliased by s_f
  u16* resV_b = (u16*)alloc((size_t)cBN3 * 512 * 2);   // }
  float* hmid_f = (float*)alloc((size_t)cBN * 512 * 4);
  float* vmid_f = (float*)alloc((size_t)cBN3 * 512 * 4);
  u16*   vmid_b = (u16*)alloc((size_t)cBN3 * 512 * 2);
  u16*   hn2_b  = (u16*)alloc((size_t)cBN * 512 * 2);
  u16*   scl_b  = (u16*)alloc((size_t)cBN * 1024 * 2);
  float* vp_f  = (float*)q_b;     // BN3*1024*4 == |q|+|k|+|vatt|
  u16*   hid_b = hv_b;            // BN*2048*2  == |hv|+|vv|
  float* s_f   = (float*)resH_b;  // BN*1024*4  == |resH|+|resV|

  // 1) weight conversions (transposed) + V input (plain)
  struct TJob { const float* s; u16* d; int K; int Nc; };
  TJob tj[9] = {
    {Wq, wq_b, 512, 2048}, {Wk, wk_b, 512, 2048}, {Wvs, wvs_b, 512, 512},
    {Wvv, wvv_b, 512, 512}, {Wo, wo_b, 512, 512}, {Wvo, wvo_b, 512, 512},
    {Wlv, wlv_b, 512, 1024}, {W1, w1_b, 1024, 2048}, {W2, w2_b, 2048, 1024}};
  for (int j = 0; j < 9; ++j) {
    int n = tj[j].K * tj[j].Nc;
    cvt_t_kernel<<<(n + 1023) / 1024, 256, 0, stream>>>(tj[j].s, tj[j].d, tj[j].K, tj[j].Nc);
  }
  cvt_kernel<<<(cBN3 * 512 + 1023) / 1024, 256, 0, stream>>>(Vin, vin_b, cBN3 * 512);

  auto gemm = [&](const u16* A, const u16* Wm, const float* bias, const float* resid,
                  float* oF, u16* oB, int M, int K, int Nc, int act) {
    dim3 g(Nc / GTN, M / GTM);
    gemm_bf16_kernel<<<g, 256, 0, stream>>>(A, Wm, bias, resid, oF, oB, M, K, Nc, act);
  };

  // 2) LN1 -> Hn (bf16)
  ln_kernel<<<cBN / 4, 128, 0, stream>>>(H, ln1g, ln1b, hn_b, cBN);
  // 3) projections
  gemm(hn_b, wq_b, bq, nullptr, nullptr, q_b, cBN, 512, 2048, 0);
  gemm(hn_b, wk_b, bk, nullptr, nullptr, k_b, cBN, 512, 2048, 0);
  gemm(hn_b, wvs_b, bvs, nullptr, nullptr, hv_b, cBN, 512, 512, 0);
  gemm(vin_b, wvv_b, nullptr, nullptr, nullptr, vv_b, cBN3, 512, 512, 0);
  pack_vattn_kernel<<<(cBN * 2048) / 1024, 256, 0, stream>>>(hv_b, vv_b, vatt_b);
  // 4) attention
  attn_kernel<<<dim3(cN / 32, cNH, cB), 128, 0, stream>>>(
      q_b, k_b, vatt_b, rbf, Dm, mask, resH_b, resV_b);
  // 5) output projections + residual
  gemm(resH_b, wo_b, bo, H, hmid_f, nullptr, cBN, 512, 512, 0);
  gemm(resV_b, wvo_b, nullptr, Vin, vmid_f, vmid_b, cBN3, 512, 512, 0);
  // 6) LN2, Vp = V_mid @ Wlv
  ln_kernel<<<cBN / 4, 128, 0, stream>>>(hmid_f, ln2g, ln2b, hn2_b, cBN);
  gemm(vmid_b, wlv_b, nullptr, nullptr, vp_f, nullptr, cBN3, 512, 1024, 0);
  // 7) scaler, FFN
  scaler_kernel<<<(cBN * 1024) / 1024, 256, 0, stream>>>(hn2_b, vp_f, scl_b);
  gemm(scl_b, w1_b, b1, nullptr, nullptr, hid_b, cBN, 1024, 2048, 1);   // silu
  gemm(hid_b, w2_b, b2, nullptr, s_f, nullptr, cBN, 2048, 1024, 0);
  // 8) final
  final_kernel<<<((cBN * 512 + cBN3 * 512)) / 1024, 256, 0, stream>>>(
      hmid_f, vmid_f, s_f, vp_f, (float*)d_out);
}
